// HierarchicalMixtureOfExperts_58806692216867
// MI455X (gfx1250) — compile-verified, matrix operand order checked by reading, not run on hardware
//
// Hierarchical MoE forward for MI455X (gfx1250), HIP + CDNA5 WMMA + TDM.
//
// Design rationale (compile-only; reasoned from MI455X specs):
//  * ~0.2 TFLOP of GEMMs vs ~184 MB of weights (fits 192 MB L2): compute path
//    must be WMMA. f16 operands (activations ~N(0,1), weights ~0.02-0.06) with
//    f32 accumulation via v_wmma_f32_16x16x32_f16.
//  * Weights are converted to f16 ONCE per launch (halves weight-side traffic,
//    removes all cvt from the GEMM hot loop). Activation producers dual-store
//    f16 copies so every GEMM consumes pre-converted f16.
//  * GEMM tiles are moved into LDS by the Tensor Data Mover (tensor_load_to_lds
//    with a 2D D#: 32x128 tile, pad_interval=16 DW / pad_amount=4 DW to produce
//    the padded 40-half row stride), double-buffered and synchronized with
//    s_wait_tensorcnt -> DMA of tile k+1 overlaps WMMAs of tile k.
//  * wave32 row kernels (LN/softmax/routers) with xor-shuffle reductions.
//  * Attention is flash-style: online softmax kept in the WMMA C-layout.

#include <hip/hip_runtime.h>
#include <hip/hip_bf16.h>

typedef _Float16 v16h __attribute__((ext_vector_type(16)));
typedef _Float16 v8h  __attribute__((ext_vector_type(8)));
typedef float    v8f  __attribute__((ext_vector_type(8)));
typedef unsigned int u32x4 __attribute__((ext_vector_type(4)));
typedef int          i32x8 __attribute__((ext_vector_type(8)));
typedef int          i32x4 __attribute__((ext_vector_type(4)));

union HFrag { v16h v; v8h h[2]; };

#if __has_builtin(__builtin_amdgcn_tensor_load_to_lds) && \
    __has_builtin(__builtin_amdgcn_s_wait_tensorcnt)
#define USE_TDM 1
#else
#define USE_TDM 0
#endif

__device__ __forceinline__ float gelu_exact(float x) {
  return 0.5f * x * (1.0f + erff(x * 0.70710678118654752440f));
}
__device__ __forceinline__ float sigmoidf_(float x) {
  return 1.0f / (1.0f + __expf(-x));
}
__device__ __forceinline__ float red_sum32(float v) {
  v += __shfl_xor(v, 1);  v += __shfl_xor(v, 2);  v += __shfl_xor(v, 4);
  v += __shfl_xor(v, 8);  v += __shfl_xor(v, 16);
  return v;
}
__device__ __forceinline__ float red_sum16(float v) {
  v += __shfl_xor(v, 1);  v += __shfl_xor(v, 2);
  v += __shfl_xor(v, 4);  v += __shfl_xor(v, 8);
  return v;
}
__device__ __forceinline__ float red_max16(float v) {
  v = fmaxf(v, __shfl_xor(v, 1)); v = fmaxf(v, __shfl_xor(v, 2));
  v = fmaxf(v, __shfl_xor(v, 4)); v = fmaxf(v, __shfl_xor(v, 8));
  return v;
}

#define GT_M 128
#define GT_N 128
#define GT_K 32
#define GLDS 40   // padded LDS row stride in halves (80B -> 16B-aligned frags)

#if USE_TDM
// Issue one TDM 2D tile load: GT_M x GT_K f16 elements from a row-major
// tensor with row length rowLen (elements) into LDS at ldsOff, with the TDM
// inserting 8 halves of padding after each 32-half row (pad_interval=16 DW,
// pad_amount=4 DW) to produce the GLDS=40 layout. D# per CDNA5 ISA ch.8.
// This toolchain's builtin takes 6 args (g0,g1,g2,g3,extra i32x8,cpol).
__device__ __forceinline__ void tdm_load_tile(const _Float16* gsrc,
                                              unsigned ldsOff, int rowLen) {
  unsigned long long ga = (unsigned long long)(uintptr_t)gsrc;
  u32x4 g0;
  g0[0] = 1u;                                   // count=1, user descriptor
  g0[1] = ldsOff;                               // lds_addr (bytes)
  g0[2] = (unsigned)ga;                         // global_addr[31:0]
  g0[3] = ((unsigned)(ga >> 32) & 0x01FFFFFFu)  // global_addr[56:32]
        | (2u << 30);                           // type = 2 ("image")
  i32x8 g1;
  g1[0] = (1 << 16)    // data_size = 2 bytes
        | (1 << 20)    // pad_enable
        | (3 << 22)    // pad_interval: 16 DWORDs (one 64B tile row)
        | (3 << 25);   // pad_amount: 4 DWORDs (8 halves)
  g1[1] = (rowLen & 0xFFFF) << 16;                       // tensor_dim0 lo16
  g1[2] = ((rowLen >> 16) & 0xFFFF) | (GT_M << 16);      // dim0 hi | dim1 lo
  g1[3] = (GT_K << 16);                                  // dim1 hi=0 | tile_dim0
  g1[4] = GT_M;                                          // tile_dim1 | tile_dim2=0
  g1[5] = rowLen;                                        // tensor_dim0_stride lo32
  g1[6] = 0;                                             // stride hi | dim1_stride lo
  g1[7] = 0;
  i32x4 z4 = {0, 0, 0, 0};
  i32x8 z8 = {0, 0, 0, 0, 0, 0, 0, 0};
  __builtin_amdgcn_tensor_load_to_lds(g0, g1, z4, z4, z8, 0);
}
#endif

// ---------------------------------------------------------------------------
// GEMM: C[M,N] = act(A[M,K] @ W[N,K]^T + bias) (+resid) (+pe), A/W in f16.
// Tile 128x128, 8 waves, 64x32 per wave (4x2 WMMA subtiles).
// ACT: 0 none, 1 exact GELU, 2 sigmoid. Dims: M,N %128==0, K %32==0.
// ---------------------------------------------------------------------------
template<int ACT, bool RES, bool ADDPE, bool S32, bool S16>
__global__ __launch_bounds__(256)
void k_gemm_h(const _Float16* __restrict__ A, const _Float16* __restrict__ W,
              const float* __restrict__ bias, const float* __restrict__ resid,
              const float* __restrict__ pe,
              float* __restrict__ C32, _Float16* __restrict__ C16,
              int M, int N, int K)
{
  __shared__ __align__(16) _Float16 As[2][GT_M * GLDS];
  __shared__ __align__(16) _Float16 Bs[2][GT_N * GLDS];

  const int tid  = threadIdx.x;
  const int lane = tid & 31;
  const int wave = tid >> 5;
  const int wm   = wave >> 2;
  const int wn   = wave & 3;
  const int hp   = lane >> 4;
  const int l16  = lane & 15;
  const int m0   = blockIdx.y * GT_M;
  const int n0   = blockIdx.x * GT_N;

  v8f acc[4][2];
  #pragma unroll
  for (int ms = 0; ms < 4; ++ms)
    #pragma unroll
    for (int ns = 0; ns < 2; ++ns) acc[ms][ns] = {};

  const int nk = K / GT_K;
  int p = 0;

#if USE_TDM
  const unsigned offA[2] = { (unsigned)(uintptr_t)&As[0][0],
                             (unsigned)(uintptr_t)&As[1][0] };
  const unsigned offB[2] = { (unsigned)(uintptr_t)&Bs[0][0],
                             (unsigned)(uintptr_t)&Bs[1][0] };
  const _Float16* Abase = A + (size_t)m0 * K;
  const _Float16* Bbase = W + (size_t)n0 * K;
  if (wave == 0) {
    tdm_load_tile(Abase, offA[0], K);
    tdm_load_tile(Bbase, offB[0], K);
  }
#endif

  for (int kt = 0; kt < nk; ++kt) {
#if USE_TDM
    if (wave == 0) {
      if (kt + 1 < nk) {
        tdm_load_tile(Abase + (kt + 1) * GT_K, offA[p ^ 1], K);
        tdm_load_tile(Bbase + (kt + 1) * GT_K, offB[p ^ 1], K);
        __builtin_amdgcn_s_wait_tensorcnt(2);  // oldest pair (buf p) done
      } else {
        __builtin_amdgcn_s_wait_tensorcnt(0);
      }
    }
    __syncthreads();
#else
    const int k0 = kt * GT_K;
    for (int i = tid; i < GT_M * 4; i += 256) {   // 128 rows x 4 x v8h
      const int row = i >> 2;
      const int c8  = (i & 3) << 3;
      *(v8h*)&As[p][row * GLDS + c8] =
          *(const v8h*)(A + (size_t)(m0 + row) * K + k0 + c8);
      *(v8h*)&Bs[p][row * GLDS + c8] =
          *(const v8h*)(W + (size_t)(n0 + row) * K + k0 + c8);
      if (k0 + GT_K < K) {
        __builtin_prefetch(A + (size_t)(m0 + row) * K + k0 + GT_K + c8, 0, 0);
        __builtin_prefetch(W + (size_t)(n0 + row) * K + k0 + GT_K + c8, 0, 0);
      }
    }
    __syncthreads();
#endif

    HFrag af[4], bf[2];
    #pragma unroll
    for (int ms = 0; ms < 4; ++ms) {
      const int r = wm * 64 + ms * 16 + l16;
      af[ms].h[0] = *(const v8h*)&As[p][r * GLDS + hp * 8];
      af[ms].h[1] = *(const v8h*)&As[p][r * GLDS + 16 + hp * 8];
    }
    #pragma unroll
    for (int ns = 0; ns < 2; ++ns) {
      const int r = wn * 32 + ns * 16 + l16;
      bf[ns].h[0] = *(const v8h*)&Bs[p][r * GLDS + hp * 8];
      bf[ns].h[1] = *(const v8h*)&Bs[p][r * GLDS + 16 + hp * 8];
    }
    #pragma unroll
    for (int ms = 0; ms < 4; ++ms)
      #pragma unroll
      for (int ns = 0; ns < 2; ++ns)
        acc[ms][ns] = __builtin_amdgcn_wmma_f32_16x16x32_f16(
            false, af[ms].v, false, bf[ns].v, (short)0, acc[ms][ns],
            false, false);
    __syncthreads();
    p ^= 1;
  }

  // epilogue: C layout vgpr r -> row = 8*hp + r, col = l16
  #pragma unroll
  for (int ms = 0; ms < 4; ++ms)
    #pragma unroll
    for (int ns = 0; ns < 2; ++ns) {
      const int col = n0 + wn * 32 + ns * 16 + l16;
      const float bv = bias[col];
      #pragma unroll
      for (int r = 0; r < 8; ++r) {
        const int row = m0 + wm * 64 + ms * 16 + hp * 8 + r;
        float v = acc[ms][ns][r] + bv;
        if (ACT == 1) v = gelu_exact(v);
        if (ACT == 2) v = sigmoidf_(v);
        if (RES)   v += resid[(size_t)row * N + col];
        if (ADDPE) v += pe[(size_t)(row & 1023) * 512 + col];
        if (S32) C32[(size_t)row * N + col] = v;
        if (S16) C16[(size_t)row * N + col] = (_Float16)v;
      }
    }
}

// ---------------------------------------------------------------------------
// Flash attention. qkv: f16 [B,S,1536] (q|k|v), out: f16 [B,S,512].
// One wave per (b, head, 16-query tile); 8 heads, hd=64, S=1024, scale 0.125.
// ---------------------------------------------------------------------------
__global__ __launch_bounds__(128)
void k_attn(const _Float16* __restrict__ qkv, _Float16* __restrict__ out)
{
  constexpr int SD = 1536;
  __shared__ __align__(16) _Float16 KsAll[4][32 * 72]; // [key][hd], pad 8
  __shared__ __align__(16) _Float16 VtAll[4][64 * 40]; // [hd][key], pad 8
  __shared__ __align__(16) _Float16 PsAll[4][16 * 40]; // [q][key],  pad 8

  const int wave = threadIdx.x >> 5;
  const int lane = threadIdx.x & 31;
  const int hp   = lane >> 4;
  const int l16  = lane & 15;
  _Float16* ks = KsAll[wave];
  _Float16* vt = VtAll[wave];
  _Float16* ps = PsAll[wave];

  const int gid = blockIdx.x * 4 + wave;   // 0..1023
  const int qb  = gid & 63;
  const int h   = (gid >> 6) & 7;
  const int b   = gid >> 9;

  // Q fragments (direct 16B global loads; qkv already f16)
  HFrag qf[2];
  const _Float16* qrow =
      qkv + (size_t)(b * 1024 + qb * 16 + l16) * SD + h * 64;
  #pragma unroll
  for (int kk = 0; kk < 2; ++kk) {
    qf[kk].h[0] = *(const v8h*)(qrow + kk * 32 + hp * 8);
    qf[kk].h[1] = *(const v8h*)(qrow + kk * 32 + 16 + hp * 8);
  }

  float mi[8], li[8];
  v8f O[4];
  #pragma unroll
  for (int r = 0; r < 8; ++r) { mi[r] = -1e30f; li[r] = 0.0f; }
  #pragma unroll
  for (int nb = 0; nb < 4; ++nb) O[nb] = {};

  for (int c = 0; c < 32; ++c) {
    const int key0 = c * 32;
    const _Float16* krow =
        qkv + (size_t)(b * 1024 + key0 + lane) * SD + 512 + h * 64;
    const _Float16* vrow =
        qkv + (size_t)(b * 1024 + key0 + lane) * SD + 1024 + h * 64;
    #pragma unroll
    for (int d = 0; d < 64; d += 8) {
      *(v8h*)&ks[lane * 72 + d] = *(const v8h*)(krow + d);
      const v8h vv = *(const v8h*)(vrow + d);
      #pragma unroll
      for (int j = 0; j < 8; ++j) vt[(d + j) * 40 + lane] = vv[j];
    }

    v8f s[2];
    #pragma unroll
    for (int t = 0; t < 2; ++t) {
      v8f sc = {};
      const int nrow = t * 16 + l16;
      #pragma unroll
      for (int kk = 0; kk < 2; ++kk) {
        HFrag kf;
        kf.h[0] = *(const v8h*)&ks[nrow * 72 + kk * 32 + hp * 8];
        kf.h[1] = *(const v8h*)&ks[nrow * 72 + kk * 32 + 16 + hp * 8];
        sc = __builtin_amdgcn_wmma_f32_16x16x32_f16(
            false, qf[kk].v, false, kf.v, (short)0, sc, false, false);
      }
      s[t] = sc;
    }

    #pragma unroll
    for (int r = 0; r < 8; ++r) {
      const float v0 = s[0][r] * 0.125f;
      const float v1 = s[1][r] * 0.125f;
      const float mx   = red_max16(fmaxf(v0, v1));
      const float mnew = fmaxf(mi[r], mx);
      const float scl  = __expf(mi[r] - mnew);
      const float p0 = __expf(v0 - mnew);
      const float p1 = __expf(v1 - mnew);
      const float rs = red_sum16(p0 + p1);
      li[r] = li[r] * scl + rs;
      mi[r] = mnew;
      #pragma unroll
      for (int nb = 0; nb < 4; ++nb) O[nb][r] = O[nb][r] * scl;
      ps[(hp * 8 + r) * 40 + l16]      = (_Float16)p0;
      ps[(hp * 8 + r) * 40 + 16 + l16] = (_Float16)p1;
    }

    HFrag pf;
    pf.h[0] = *(const v8h*)&ps[l16 * 40 + hp * 8];
    pf.h[1] = *(const v8h*)&ps[l16 * 40 + 16 + hp * 8];
    #pragma unroll
    for (int nb = 0; nb < 4; ++nb) {
      HFrag vf;
      const int nrow = nb * 16 + l16;
      vf.h[0] = *(const v8h*)&vt[nrow * 40 + hp * 8];
      vf.h[1] = *(const v8h*)&vt[nrow * 40 + 16 + hp * 8];
      O[nb] = __builtin_amdgcn_wmma_f32_16x16x32_f16(
          false, pf.v, false, vf.v, (short)0, O[nb], false, false);
    }
  }

  #pragma unroll
  for (int r = 0; r < 8; ++r) {
    const float inv = 1.0f / li[r];
    const int row = b * 1024 + qb * 16 + hp * 8 + r;
    #pragma unroll
    for (int nb = 0; nb < 4; ++nb)
      out[(size_t)row * 512 + h * 64 + nb * 16 + l16] =
          (_Float16)(O[nb][r] * inv);
  }
}

// ---------------------------------------------------------------------------
// Row kernels (warp-per-row, D = 512)
// ---------------------------------------------------------------------------
__global__ __launch_bounds__(256)
void k_ln(const float* __restrict__ a, const float* __restrict__ g,
          const float* __restrict__ bta, float* __restrict__ o32,
          _Float16* __restrict__ o16)
{
  const int warp = threadIdx.x >> 5, lane = threadIdx.x & 31;
  const size_t base = ((size_t)blockIdx.x * 8 + warp) * 512;
  float v[16], s = 0.f, s2 = 0.f;
  #pragma unroll
  for (int i = 0; i < 16; ++i) {
    const float t = a[base + lane + i * 32];
    v[i] = t; s += t; s2 += t * t;
  }
  s = red_sum32(s); s2 = red_sum32(s2);
  const float mean = s * (1.0f / 512.0f);
  const float var  = s2 * (1.0f / 512.0f) - mean * mean;
  const float rs   = rsqrtf(var + 1e-5f);
  #pragma unroll
  for (int i = 0; i < 16; ++i) {
    const int c = lane + i * 32;
    const float t = (v[i] - mean) * rs * g[c] + bta[c];
    if (o32) o32[base + c] = t;
    if (o16) o16[base + c] = (_Float16)t;
  }
}

// accum[row] (+)= scores[row,sub] * LN(y[row] + x2[row])
__global__ __launch_bounds__(256)
void k_ln_scale_acc(const float* __restrict__ y, const float* __restrict__ x2,
                    const float* __restrict__ g, const float* __restrict__ bta,
                    const float* __restrict__ sc, int sub,
                    float* __restrict__ acc, int init)
{
  const int warp = threadIdx.x >> 5, lane = threadIdx.x & 31;
  const size_t row  = (size_t)blockIdx.x * 8 + warp;
  const size_t base = row * 512;
  float v[16], s = 0.f, s2 = 0.f;
  #pragma unroll
  for (int i = 0; i < 16; ++i) {
    const float t = y[base + lane + i * 32] + x2[base + lane + i * 32];
    v[i] = t; s += t; s2 += t * t;
  }
  s = red_sum32(s); s2 = red_sum32(s2);
  const float mean = s * (1.0f / 512.0f);
  const float var  = s2 * (1.0f / 512.0f) - mean * mean;
  const float rs   = rsqrtf(var + 1e-5f);
  const float w    = sc[row * 4 + sub];
  #pragma unroll
  for (int i = 0; i < 16; ++i) {
    const int c = lane + i * 32;
    const float t = ((v[i] - mean) * rs * g[c] + bta[c]) * w;
    acc[base + c] = init ? t : (acc[base + c] + t);
  }
}

// out[row] (+)= wts[row,e] * LN(gate[row]*acc[row] + x2[row])
__global__ __launch_bounds__(256)
void k_final_acc(const float* __restrict__ acc, const float* __restrict__ gate,
                 const float* __restrict__ x2, const float* __restrict__ g,
                 const float* __restrict__ bta, const float* __restrict__ wts,
                 int e, float* __restrict__ out, int init)
{
  const int warp = threadIdx.x >> 5, lane = threadIdx.x & 31;
  const size_t row  = (size_t)blockIdx.x * 8 + warp;
  const size_t base = row * 512;
  float v[16], s = 0.f, s2 = 0.f;
  #pragma unroll
  for (int i = 0; i < 16; ++i) {
    const size_t idx = base + lane + i * 32;
    const float t = gate[idx] * acc[idx] + x2[idx];
    v[i] = t; s += t; s2 += t * t;
  }
  s = red_sum32(s); s2 = red_sum32(s2);
  const float mean = s * (1.0f / 512.0f);
  const float var  = s2 * (1.0f / 512.0f) - mean * mean;
  const float rs   = rsqrtf(var + 1e-5f);
  const float w    = wts[row * 4 + e];
  #pragma unroll
  for (int i = 0; i < 16; ++i) {
    const int c = lane + i * 32;
    const float t = ((v[i] - mean) * rs * g[c] + bta[c]) * w;
    out[base + c] = init ? t : (out[base + c] + t);
  }
}

// outer router: softmax over 4 -> top-2 -> renormalized dense weights
__global__ __launch_bounds__(256)
void k_router(const float* __restrict__ x, const float* __restrict__ rW,
              const float* __restrict__ rb, float* __restrict__ w)
{
  const int warp = threadIdx.x >> 5, lane = threadIdx.x & 31;
  const size_t row = (size_t)blockIdx.x * 8 + warp;
  const float* px = x + row * 512;
  float lg[4];
  #pragma unroll
  for (int e = 0; e < 4; ++e) {
    float s = 0.f;
    for (int d = lane; d < 512; d += 32) s += px[d] * rW[e * 512 + d];
    lg[e] = red_sum32(s) + rb[e];
  }
  const float mx = fmaxf(fmaxf(lg[0], lg[1]), fmaxf(lg[2], lg[3]));
  float p[4], psum = 0.f;
  #pragma unroll
  for (int e = 0; e < 4; ++e) { p[e] = __expf(lg[e] - mx); psum += p[e]; }
  #pragma unroll
  for (int e = 0; e < 4; ++e) p[e] /= psum;
  int i1 = 0; float v1 = p[0];
  #pragma unroll
  for (int e = 1; e < 4; ++e) if (p[e] > v1) { v1 = p[e]; i1 = e; }
  int i2 = -1; float v2 = -1.f;
  #pragma unroll
  for (int e = 0; e < 4; ++e)
    if (e != i1 && p[e] > v2) { v2 = p[e]; i2 = e; }
  const float inv = 1.0f / (v1 + v2);
  if (lane == 0)
    #pragma unroll
    for (int e = 0; e < 4; ++e)
      w[row * 4 + e] = (e == i1) ? v1 * inv : ((e == i2) ? v2 * inv : 0.0f);
}

// sub-router: softmax((x2 @ srW^T + srb) / temp) over 4
__global__ __launch_bounds__(256)
void k_subrouter(const float* __restrict__ x2, const float* __restrict__ srW,
                 const float* __restrict__ srb, const float* __restrict__ tp,
                 float* __restrict__ sc)
{
  const int warp = threadIdx.x >> 5, lane = threadIdx.x & 31;
  const size_t row = (size_t)blockIdx.x * 8 + warp;
  const float* px = x2 + row * 512;
  const float invT = 1.0f / tp[0];
  float lg[4];
  #pragma unroll
  for (int k = 0; k < 4; ++k) {
    float s = 0.f;
    for (int d = lane; d < 512; d += 32) s += px[d] * srW[k * 512 + d];
    lg[k] = (red_sum32(s) + srb[k]) * invT;
  }
  const float mx = fmaxf(fmaxf(lg[0], lg[1]), fmaxf(lg[2], lg[3]));
  float p[4], psum = 0.f;
  #pragma unroll
  for (int k = 0; k < 4; ++k) { p[k] = __expf(lg[k] - mx); psum += p[k]; }
  if (lane == 0)
    #pragma unroll
    for (int k = 0; k < 4; ++k) sc[row * 4 + k] = p[k] / psum;
}

// f32 -> f16 bulk convert (8 elems/thread; counts divisible by 2048)
__global__ __launch_bounds__(256)
void k_f2h(const float* __restrict__ s, _Float16* __restrict__ d)
{
  const size_t i = ((size_t)blockIdx.x * 256 + threadIdx.x) * 8;
  const float4 a = *(const float4*)(s + i);
  const float4 b = *(const float4*)(s + i + 4);
  v8h o;
  o[0]=(_Float16)a.x; o[1]=(_Float16)a.y; o[2]=(_Float16)a.z; o[3]=(_Float16)a.w;
  o[4]=(_Float16)b.x; o[5]=(_Float16)b.y; o[6]=(_Float16)b.z; o[7]=(_Float16)b.w;
  *(v8h*)(d + i) = o;
}

// ---------------------------------------------------------------------------
// Host orchestration. Input order = setup_inputs() dict, depth-first:
// 0:x 1:router_W 2:router_b 3:pe 4:ln0_g 5:ln0_b 6:proj1_W 7:proj1_b
// 8:proj2_W 9:proj2_b 10:attn_in_W 11:attn_in_b 12:attn_out_W 13:attn_out_b
// 14:attn_ln_g 15:attn_ln_b 16:sr_W 17:sr_b 18:temp 19:gate_ln_g 20:gate_ln_b
// 21:gate1_W 22:gate1_b 23:gate2_W 24:gate2_b 25:sub_fc1_W 26:sub_fc1_b
// 27:sub_fc2_W 28:sub_fc2_b 29:sub_ln_g 30:sub_ln_b 31:ln_g 32:ln_b
// ---------------------------------------------------------------------------
extern "C" void kernel_launch(void* const* d_in, const int* in_sizes, int n_in,
                              void* d_out, int out_size, void* d_ws, size_t ws_size,
                              hipStream_t stream)
{
  (void)in_sizes; (void)n_in; (void)out_size; (void)ws_size;
  constexpr int Bc = 2, Sc = 1024, Dc = 512, Ec = 4, SUBc = 4;
  constexpr int M = Bc * Sc;
  const size_t BSD = (size_t)M * Dc;

  const float* X   = (const float*)d_in[0];
  const float* RW  = (const float*)d_in[1];
  const float* RB  = (const float*)d_in[2];
  const float* PE  = (const float*)d_in[3];
  const float* LN0G = (const float*)d_in[4];
  const float* LN0B = (const float*)d_in[5];
  const float* P1W = (const float*)d_in[6];
  const float* P1B = (const float*)d_in[7];
  const float* P2W = (const float*)d_in[8];
  const float* P2B = (const float*)d_in[9];
  const float* AIW = (const float*)d_in[10];
  const float* AIB = (const float*)d_in[11];
  const float* AOW = (const float*)d_in[12];
  const float* AOB = (const float*)d_in[13];
  const float* ALG = (const float*)d_in[14];
  const float* ALB = (const float*)d_in[15];
  const float* SRW = (const float*)d_in[16];
  const float* SRB = (const float*)d_in[17];
  const float* TMP = (const float*)d_in[18];
  const float* GLG = (const float*)d_in[19];
  const float* GLB = (const float*)d_in[20];
  const float* G1W = (const float*)d_in[21];
  const float* G1B = (const float*)d_in[22];
  const float* G2W = (const float*)d_in[23];
  const float* G2B = (const float*)d_in[24];
  const float* F1W = (const float*)d_in[25];
  const float* F1B = (const float*)d_in[26];
  const float* F2W = (const float*)d_in[27];
  const float* F2B = (const float*)d_in[28];
  const float* SLG = (const float*)d_in[29];
  const float* SLB = (const float*)d_in[30];
  const float* LNG = (const float*)d_in[31];
  const float* LNB = (const float*)d_in[32];
  float* out = (float*)d_out;

  // f32 scratch
  float* fp    = (float*)d_ws;
  float* t_x   = fp;            fp += BSD;   // expert working x (post proj/pe)
  float* t_y   = fp;            fp += BSD;   // AO out / fc2 out
  float* t_x2  = fp;            fp += BSD;   // post-attn LN x
  float* t_gat = fp;            fp += BSD;   // gate
  float* t_acc = fp;            fp += BSD;   // sub-expert weighted sum
  float* t_sc  = fp;            fp += (size_t)M * SUBc;
  float* t_w   = fp;            fp += (size_t)M * Ec;
  // f16 activations
  _Float16* hq = (_Float16*)fp;
  _Float16* a_ln16  = hq;       hq += BSD;      // LN outputs feeding GEMMs
  _Float16* a_big16 = hq;       hq += 4 * BSD;  // proj1/gate1/fc1 outputs
  _Float16* a_x16   = hq;       hq += BSD;      // x (qkv GEMM input)
  _Float16* a_qkv16 = hq;       hq += 3 * BSD;
  _Float16* a_att16 = hq;       hq += BSD;
  _Float16* a_x216  = hq;       hq += BSD;      // x2 (fc1 input)
  // f16 weights (converted once per launch)
  const size_t DD = (size_t)Dc * Dc;
  _Float16* hP1 = hq;  hq += (size_t)Ec * 2 * DD;
  _Float16* hP2 = hq;  hq += (size_t)Ec * 2 * DD;
  _Float16* hAI = hq;  hq += (size_t)Ec * 3 * DD;
  _Float16* hAO = hq;  hq += (size_t)Ec * DD;
  _Float16* hG1 = hq;  hq += (size_t)Ec * DD;
  _Float16* hG2 = hq;  hq += (size_t)Ec * DD;
  _Float16* hF1 = hq;  hq += (size_t)Ec * SUBc * 4 * DD;
  _Float16* hF2 = hq;  hq += (size_t)Ec * SUBc * 4 * DD;

  auto f2h = [&](const float* s, _Float16* d, size_t n) {
    k_f2h<<<(unsigned)(n / 2048), 256, 0, stream>>>(s, d);
  };
  f2h(P1W, hP1, (size_t)Ec * 2 * DD);
  f2h(P2W, hP2, (size_t)Ec * 2 * DD);
  f2h(AIW, hAI, (size_t)Ec * 3 * DD);
  f2h(AOW, hAO, (size_t)Ec * DD);
  f2h(G1W, hG1, (size_t)Ec * DD);
  f2h(G2W, hG2, (size_t)Ec * DD);
  f2h(F1W, hF1, (size_t)Ec * SUBc * 4 * DD);
  f2h(F2W, hF2, (size_t)Ec * SUBc * 4 * DD);

  k_router<<<M / 8, 256, 0, stream>>>(X, RW, RB, t_w);

  for (int e = 0; e < Ec; ++e) {
    // input_proj: x = X + proj2(GELU(proj1(LN(X)))) + pe
    k_ln<<<M / 8, 256, 0, stream>>>(X, LN0G + e * Dc, LN0B + e * Dc,
                                    nullptr, a_ln16);
    k_gemm_h<1,false,false,false,true><<<dim3(8, 16), 256, 0, stream>>>(
        a_ln16, hP1 + (size_t)e * 2 * DD, P1B + e * 2 * Dc,
        nullptr, nullptr, nullptr, a_big16, M, 2 * Dc, Dc);
    k_gemm_h<0,true,true,true,true><<<dim3(4, 16), 256, 0, stream>>>(
        a_big16, hP2 + (size_t)e * 2 * DD, P2B + e * Dc,
        X, PE + (size_t)e * Sc * Dc, t_x, a_x16, M, Dc, 2 * Dc);
    // attention: qkv (f16-only), flash attention, out-proj + residual
    k_gemm_h<0,false,false,false,true><<<dim3(12, 16), 256, 0, stream>>>(
        a_x16, hAI + (size_t)e * 3 * DD, AIB + e * 3 * Dc,
        nullptr, nullptr, nullptr, a_qkv16, M, 3 * Dc, Dc);
    k_attn<<<256, 128, 0, stream>>>(a_qkv16, a_att16);
    k_gemm_h<0,true,false,true,false><<<dim3(4, 16), 256, 0, stream>>>(
        a_att16, hAO + (size_t)e * DD, AOB + e * Dc,
        t_x, nullptr, t_y, nullptr, M, Dc, Dc);
    k_ln<<<M / 8, 256, 0, stream>>>(t_y, ALG + e * Dc, ALB + e * Dc,
                                    t_x2, a_x216);
    // sub-router scores (learned temperature)
    k_subrouter<<<M / 8, 256, 0, stream>>>(t_x2, SRW + (size_t)e * SUBc * Dc,
                                           SRB + e * SUBc, TMP + e, t_sc);
    // gate = sigmoid(gate2(GELU(gate1(LN(x2)))))
    k_ln<<<M / 8, 256, 0, stream>>>(t_x2, GLG + e * Dc, GLB + e * Dc,
                                    nullptr, a_ln16);
    k_gemm_h<1,false,false,false,true><<<dim3(4, 16), 256, 0, stream>>>(
        a_ln16, hG1 + (size_t)e * DD, G1B + e * Dc,
        nullptr, nullptr, nullptr, a_big16, M, Dc, Dc);
    k_gemm_h<2,false,false,true,false><<<dim3(4, 16), 256, 0, stream>>>(
        a_big16, hG2 + (size_t)e * DD, G2B + e * Dc,
        nullptr, nullptr, t_gat, nullptr, M, Dc, Dc);
    // sub experts: acc = sum_k scores[:,k] * LN(fc2(GELU(fc1(x2))) + x2)
    for (int k = 0; k < SUBc; ++k) {
      const size_t ek = (size_t)(e * SUBc + k);
      k_gemm_h<1,false,false,false,true><<<dim3(16, 16), 256, 0, stream>>>(
          a_x216, hF1 + ek * 4 * DD, F1B + ek * 4 * Dc,
          nullptr, nullptr, nullptr, a_big16, M, 4 * Dc, Dc);
      k_gemm_h<0,false,false,true,false><<<dim3(4, 16), 256, 0, stream>>>(
          a_big16, hF2 + ek * 4 * DD, F2B + ek * Dc,
          nullptr, nullptr, t_y, nullptr, M, Dc, 4 * Dc);
      k_ln_scale_acc<<<M / 8, 256, 0, stream>>>(
          t_y, t_x2, SLG + ek * Dc, SLB + ek * Dc, t_sc, k, t_acc,
          (k == 0) ? 1 : 0);
    }
    // out (+)= w[:,e] * LN(gate*acc + x2)
    k_final_acc<<<M / 8, 256, 0, stream>>>(
        t_acc, t_gat, t_x2, LNG + e * Dc, LNB + e * Dc, t_w, e, out,
        (e == 0) ? 1 : 0);
  }
}
